// RWKV7_695784702707
// MI455X (gfx1250) — matrix operand order repaired and checked by reading, not compiled, not run on hardware
//
#include <hip/hip_runtime.h>
#include <hip/hip_bf16.h>

// ---------------------------------------------------------------------------
// RWKV7 forward for MI455X (gfx1250, wave32, WMMA)
// Weights are re-packed once per call to f16 in WMMA tile order; GEMMs run
// v_wmma_f32_16x16x32_f16 (f16 mul, f32 accumulate) with a double-buffered
// LDS pipeline (one barrier per K-step, global loads overlapped with WMMA).
// ---------------------------------------------------------------------------

#define CDIM   768
#define TT     512
#define BB     2
#define NTOK   (BB * TT)          // 1024
#define NHD    12
#define HSZ    64
#define VOCAB  32000
#define DFFN   3072
#define NC     (NTOK * CDIM)      // 786432 floats per (B,T,C) buffer

typedef __attribute__((ext_vector_type(16))) _Float16 v16h;
typedef __attribute__((ext_vector_type(8)))  float    v8f;

// ---------------------------------------------------------------------------
// Weight pre-pack: fp32 [K][N] row-major -> f16 tiles
// layout: [ntile][ktile][n_local(128)][k_local(32)], zero-padded in N.
// ---------------------------------------------------------------------------
__global__ void pack_w_kernel(_Float16* __restrict__ dst, const float* __restrict__ src,
                              int N, int ktiles, long total) {
  long i = (long)blockIdx.x * 256 + threadIdx.x;
  if (i >= total) return;
  int  within = (int)(i & 4095);
  long tile   = i >> 12;
  int  kt     = (int)(tile % ktiles);
  long nt     = tile / ktiles;
  int  n_l    = within >> 5;
  int  k_l    = within & 31;
  long n      = nt * 128 + n_l;
  int  k      = kt * 32 + k_l;
  dst[i] = (_Float16)((n < N) ? src[(size_t)k * N + n] : 0.f);
}

// ---------------------------------------------------------------------------
// WMMA GEMM: C[M,N] = act(A[M,K] @ W[K,N]) (+ Res if non-null)
// A,C fp32 row-major; B pre-packed f16 tiles. M%64==0, K%32==0 required.
// Block = 256 threads (8 waves), tile 64x128x32, double-buffered LDS.
// Wave (wm 0..3, wn 0..1) owns a 16x64 slab: 4 accumulators, A-fragment
// reused 4x per K-step. ACT: 0=none 1=tanh 2=sigmoid 3=relu^2.
// ---------------------------------------------------------------------------
template<int ACT>
__global__ __launch_bounds__(256)
void gemm_wmma(const float* __restrict__ A, const _Float16* __restrict__ Bp,
               float* __restrict__ Cmat, const float* __restrict__ Res,
               int M, int N, int K) {
  __shared__ __align__(16) _Float16 As[2 * 64 * 32];    // [buf][m][k]
  __shared__ __align__(16) _Float16 Bs[2 * 128 * 32];   // [buf][n][k] packed

  const int tid  = threadIdx.x;
  const int m0   = blockIdx.y * 64;
  const int n0   = blockIdx.x * 128;
  const int wave = tid >> 5;
  const int lane = tid & 31;
  const int wm   = wave & 3;      // M sub-tile (16 rows)
  const int wn   = wave >> 2;     // N half (64 cols)
  const int lm   = lane & 15;
  const int lhi  = lane >> 4;
  const int nk   = K >> 5;

  const size_t btile_base = (size_t)blockIdx.x * nk * 4096;

  v8f acc[4] = {v8f{}, v8f{}, v8f{}, v8f{}};

  // staging indices (constant across K loop)
  const int aidx = tid * 8;            // 8 elems/thread of 64x32 A tile
  const int arow = aidx >> 5, acol = aidx & 31;
  const int bidx = tid * 16;           // 16 halves/thread of 128x32 B tile

  // register staging for the pipelined tile
  float4 ra0, ra1;
  uint4  rb0, rb1;

  auto load_tile = [&](int kt) {
    const float* src = A + (size_t)(m0 + arow) * K + (kt << 5) + acol;
    ra0 = *(const float4*)(src);
    ra1 = *(const float4*)(src + 4);
    const _Float16* bsrc = Bp + btile_base + (size_t)kt * 4096;
    rb0 = *(const uint4*)&bsrc[bidx];
    rb1 = *(const uint4*)&bsrc[bidx + 8];
  };
  auto store_tile = [&](int buf) {
    union { _Float16 h[8]; uint4 u; } ap;
    ap.h[0] = (_Float16)ra0.x; ap.h[1] = (_Float16)ra0.y;
    ap.h[2] = (_Float16)ra0.z; ap.h[3] = (_Float16)ra0.w;
    ap.h[4] = (_Float16)ra1.x; ap.h[5] = (_Float16)ra1.y;
    ap.h[6] = (_Float16)ra1.z; ap.h[7] = (_Float16)ra1.w;
    *(uint4*)&As[buf * 2048 + aidx]     = ap.u;
    *(uint4*)&Bs[buf * 4096 + bidx]     = rb0;
    *(uint4*)&Bs[buf * 4096 + bidx + 8] = rb1;
  };

  load_tile(0);
  store_tile(0);
  __syncthreads();

  for (int kt = 0; kt < nk; kt++) {
    const int  cur  = kt & 1;
    const bool more = (kt + 1) < nk;
    if (more) load_tile(kt + 1);      // global loads issue before WMMAs

    // --- A fragment (16x32): k = 16*lhi + s, same contiguous order as B ---
    union { v16h v; unsigned u[8]; } af;
    {
      const _Float16* ar = &As[cur * 2048 + (wm * 16 + lm) * 32];
      #pragma unroll
      for (int vg = 0; vg < 8; vg++) {
        int kk2 = (vg << 1) + lhi * 16;
        af.u[vg] = *(const unsigned*)(ar + kk2);
      }
    }
    // --- 4 B fragments + 4 WMMAs, A reused ---
    #pragma unroll
    for (int nt = 0; nt < 4; nt++) {
      union { v16h v; uint4 u[2]; } bf;
      const _Float16* bc = &Bs[cur * 4096 + (wn * 64 + nt * 16 + lm) * 32 + lhi * 16];
      bf.u[0] = *(const uint4*)(bc);
      bf.u[1] = *(const uint4*)(bc + 8);
      acc[nt] = __builtin_amdgcn_wmma_f32_16x16x32_f16(false, af.v, false, bf.v,
                                                       (short)0, acc[nt], false, false);
    }

    if (more) store_tile(1 - cur);    // write other buffer (no race with cur)
    __syncthreads();                  // single barrier per K-step
  }

  // --- epilogue: VGPR j -> row j (+8 for lanes 16..31), col = lm ---
  #pragma unroll
  for (int nt = 0; nt < 4; nt++) {
    int colg = n0 + wn * 64 + nt * 16 + lm;
    if (colg >= N) continue;
    #pragma unroll
    for (int j = 0; j < 8; j++) {
      int rowg = m0 + wm * 16 + ((lane < 16) ? j : j + 8);
      float val = acc[nt][j];
      if (ACT == 1)      val = tanhf(val);
      else if (ACT == 2) val = 1.f / (1.f + expf(-val));
      else if (ACT == 3) { float rr = fmaxf(val, 0.f); val = rr * rr; }
      size_t o = (size_t)rowg * N + colg;
      if (Res) val += Res[o];
      Cmat[o] = val;
    }
  }
}

// ---------------------------------------------------------------------------
// Elementwise / normalization kernels
// ---------------------------------------------------------------------------

__global__ void embed_kernel(const int* __restrict__ ids,
                             const float* __restrict__ emb,
                             float* __restrict__ x) {
  int row = blockIdx.x;
  int id = ids[row];
  for (int c = threadIdx.x; c < CDIM; c += 256)
    x[(size_t)row * CDIM + c] = emb[(size_t)id * CDIM + c];
}

// LayerNorm over C=768 per row; safe in-place (dst may == src).
__global__ void ln_kernel(float* __restrict__ dst, const float* __restrict__ src,
                          const float* __restrict__ w, const float* __restrict__ b,
                          float eps) {
  __shared__ float red[256];
  int row = blockIdx.x, tid = threadIdx.x;
  const float* s = src + (size_t)row * CDIM;
  float v0 = s[tid], v1 = s[tid + 256], v2 = s[tid + 512];
  red[tid] = v0 + v1 + v2;
  __syncthreads();
  for (int off = 128; off > 0; off >>= 1) {
    if (tid < off) red[tid] += red[tid + off];
    __syncthreads();
  }
  float mean = red[0] * (1.f / CDIM);
  __syncthreads();
  float d0 = v0 - mean, d1 = v1 - mean, d2 = v2 - mean;
  red[tid] = d0 * d0 + d1 * d1 + d2 * d2;
  __syncthreads();
  for (int off = 128; off > 0; off >>= 1) {
    if (tid < off) red[tid] += red[tid + off];
    __syncthreads();
  }
  float rstd = rsqrtf(red[0] * (1.f / CDIM) + eps);
  float* d = dst + (size_t)row * CDIM;
  d[tid]       = d0 * rstd * w[tid]       + b[tid];
  d[tid + 256] = d1 * rstd * w[tid + 256] + b[tid + 256];
  d[tid + 512] = d2 * rstd * w[tid + 512] + b[tid + 512];
}

// token-shift mix for 6 targets: out_i = cur + (prev - cur) * mix_i
__global__ void mix6_kernel(const float* __restrict__ xn,
                            const float* mr, const float* mw, const float* mk,
                            const float* mv, const float* ma, const float* mg,
                            float* dr, float* dw, float* dk,
                            float* dv, float* da, float* dg) {
  int row = blockIdx.x;
  int t = row % TT;
  for (int c = threadIdx.x; c < CDIM; c += 256) {
    size_t i = (size_t)row * CDIM + c;
    float cur = xn[i];
    float prev = t ? xn[i - CDIM] : 0.f;
    float dx = prev - cur;
    dr[i] = cur + dx * mr[c];
    dw[i] = cur + dx * mw[c];
    dk[i] = cur + dx * mk[c];
    dv[i] = cur + dx * mv[c];
    da[i] = cur + dx * ma[c];
    dg[i] = cur + dx * mg[c];
  }
}

__global__ void mix1_kernel(const float* __restrict__ xn,
                            const float* mk, float* dk) {
  int row = blockIdx.x;
  int t = row % TT;
  for (int c = threadIdx.x; c < CDIM; c += 256) {
    size_t i = (size_t)row * CDIM + c;
    float cur = xn[i];
    float prev = t ? xn[i - CDIM] : 0.f;
    dk[i] = cur + (prev - cur) * mk[c];
  }
}

// w_dec = exp(-exp(-softplus(-(w0 + t2)) - 0.5))
__global__ void wdecay_kernel(float* __restrict__ wdec, const float* __restrict__ t2,
                              const float* __restrict__ w0, int n) {
  int idx = blockIdx.x * 256 + threadIdx.x;
  if (idx >= n) return;
  int c = idx % CDIM;
  float u = w0[c] + t2[idx];
  float sp = log1pf(expf(-u));           // softplus(-u)
  wdec[idx] = expf(-expf(-sp - 0.5f));
}

// v = v + (v_first - v) * sigmoid(v0 + t2)
__global__ void vmix_kernel(float* __restrict__ v, const float* __restrict__ vfirst,
                            const float* __restrict__ t2, const float* __restrict__ v0,
                            int n) {
  int idx = blockIdx.x * 256 + threadIdx.x;
  if (idx >= n) return;
  int c = idx % CDIM;
  float s = 1.f / (1.f + expf(-(v0[c] + t2[idx])));
  float vv = v[idx];
  v[idx] = vv + (vfirst[idx] - vv) * s;
}

// a = sigmoid(a0 + t2)
__global__ void asig_kernel(float* __restrict__ a, const float* __restrict__ t2,
                            const float* __restrict__ a0, int n) {
  int idx = blockIdx.x * 256 + threadIdx.x;
  if (idx >= n) return;
  int c = idx % CDIM;
  a[idx] = 1.f / (1.f + expf(-(a0[c] + t2[idx])));
}

__global__ void copy_kernel(float* __restrict__ d, const float* __restrict__ s, int n) {
  int idx = blockIdx.x * 256 + threadIdx.x;
  if (idx < n) d[idx] = s[idx];
}

// per-(b,t,h): kk = normalize(k*k_k); k *= (1 + (a-1)*k_a)
__global__ void kk_kernel(float* __restrict__ kk, float* __restrict__ k,
                          const float* __restrict__ a,
                          const float* __restrict__ k_k, const float* __restrict__ k_a) {
  __shared__ float red[HSZ];
  int bi = blockIdx.x;
  int h = bi % NHD, row = bi / NHD;
  int i = threadIdx.x;
  int c = h * HSZ + i;
  size_t idx = (size_t)row * CDIM + c;
  float kv = k[idx];
  float kkraw = kv * k_k[c];
  red[i] = kkraw * kkraw;
  __syncthreads();
  for (int off = 32; off > 0; off >>= 1) {
    if (i < off) red[i] += red[i + off];
    __syncthreads();
  }
  float denom = fmaxf(sqrtf(red[0]), 1e-12f);
  kk[idx] = kkraw / denom;
  float av = a[idx];
  k[idx] = kv * (1.f + (av - 1.f) * k_a[c]);
}

// Sequential WKV7 scan. One block per (b,h); 64 threads, thread i owns
// state row i (64 floats kept in registers; loops fully unrolled).
__global__ __launch_bounds__(64)
void wkv_kernel(float* __restrict__ y, const float* __restrict__ r,
                const float* __restrict__ w, const float* __restrict__ k,
                const float* __restrict__ v, const float* __restrict__ kk,
                const float* __restrict__ a) {
  __shared__ float rs[HSZ], ws[HSZ], ks[HSZ], vs[HSZ], as_[HSZ], bs[HSZ];
  int bh = blockIdx.x;
  int h = bh % NHD, b = bh / NHD;
  int i = threadIdx.x;
  float st[HSZ];
  #pragma unroll
  for (int j = 0; j < HSZ; j++) st[j] = 0.f;

  for (int t = 0; t < TT; t++) {
    size_t idx = ((size_t)(b * TT + t)) * CDIM + h * HSZ + i;
    rs[i] = r[idx]; ws[i] = w[idx]; ks[i] = k[idx]; vs[i] = v[idx];
    float kkv = kk[idx];
    as_[i] = -kkv;                  // a_t = -kk
    bs[i]  = kkv * a[idx];          // b_t = kk * sigmoid_a
    __syncthreads();

    float sa = 0.f;
    #pragma unroll
    for (int j = 0; j < HSZ; j++) sa += st[j] * as_[j];
    float vi = vs[i];
    float yv = 0.f;
    #pragma unroll
    for (int j = 0; j < HSZ; j++) {
      st[j] = st[j] * ws[j] + sa * bs[j] + vi * ks[j];
      yv += st[j] * rs[j];
    }
    y[idx] = yv;
    __syncthreads();
  }
}

// out = (GroupNorm(y) + ((r*k*r_k).sum * v)) * g   per (b,t,h)
__global__ void gnbonus_kernel(float* __restrict__ out, const float* __restrict__ y,
                               const float* __restrict__ r, const float* __restrict__ k,
                               const float* __restrict__ v, const float* __restrict__ g,
                               const float* __restrict__ lnxw, const float* __restrict__ lnxb,
                               const float* __restrict__ r_k) {
  __shared__ float s_sum[HSZ], s_sq[HSZ], s_rk[HSZ];
  int bi = blockIdx.x;
  int h = bi % NHD, row = bi / NHD;
  int i = threadIdx.x;
  int c = h * HSZ + i;
  size_t idx = (size_t)row * CDIM + c;
  float yv = y[idx];
  s_sum[i] = yv;
  s_sq[i]  = yv * yv;
  s_rk[i]  = r[idx] * k[idx] * r_k[c];
  __syncthreads();
  for (int off = 32; off > 0; off >>= 1) {
    if (i < off) {
      s_sum[i] += s_sum[i + off];
      s_sq[i]  += s_sq[i + off];
      s_rk[i]  += s_rk[i + off];
    }
    __syncthreads();
  }
  float m   = s_sum[0] * (1.f / HSZ);
  float var = s_sq[0] * (1.f / HSZ) - m * m;
  float gn  = (yv - m) * rsqrtf(var + 64e-5f) * lnxw[c] + lnxb[c];
  float bonus = s_rk[0] * v[idx];
  out[idx] = (gn + bonus) * g[idx];
}

// ---------------------------------------------------------------------------
// Host-side orchestration
// ---------------------------------------------------------------------------

enum { P_LN1W = 0, P_LN1B, P_LN2W, P_LN2B,
       P_XR, P_XW, P_XK, P_XV, P_XA, P_XG,
       P_WR, P_WK, P_WV, P_WO,
       P_W0, P_W1, P_W2,
       P_A0, P_A1, P_A2,
       P_V0, P_V1, P_V2,
       P_G1, P_G2,
       P_KKP, P_KAP, P_RK,
       P_LNXW, P_LNXB,
       P_XKF, P_WKF, P_WVF,
       P_LN0W, P_LN0B };

static void launch_gemm(int act, const float* A, const _Float16* Bp, float* C,
                        const float* Res, int M, int N, int K, hipStream_t s) {
  dim3 g((N + 127) / 128, M / 64), b(256);
  switch (act) {
    case 1: gemm_wmma<1><<<g, b, 0, s>>>(A, Bp, C, Res, M, N, K); break;
    case 2: gemm_wmma<2><<<g, b, 0, s>>>(A, Bp, C, Res, M, N, K); break;
    case 3: gemm_wmma<3><<<g, b, 0, s>>>(A, Bp, C, Res, M, N, K); break;
    default: gemm_wmma<0><<<g, b, 0, s>>>(A, Bp, C, Res, M, N, K); break;
  }
}

extern "C" void kernel_launch(void* const* d_in, const int* in_sizes, int n_in,
                              void* d_out, int out_size, void* d_ws, size_t ws_size,
                              hipStream_t stream) {
  (void)in_sizes; (void)n_in; (void)ws_size; (void)out_size;

  const int*   ids = (const int*)d_in[0];
  const float* emb = (const float*)d_in[1];
  auto LP = [&](int layer, int off) -> const float* {
    int base = (layer == 0) ? 2 : (37 + (layer - 1) * 33);
    return (const float*)d_in[base + off];
  };
  const float* lno_w = (const float*)d_in[202];
  const float* lno_b = (const float*)d_in[203];
  const float* headW = (const float*)d_in[204];

  float* W = (float*)d_ws;
  float* x      = W + 0 * (size_t)NC;
  float* xn     = W + 1 * (size_t)NC;
  float* xr     = W + 2 * (size_t)NC;
  float* xw     = W + 3 * (size_t)NC;
  float* xk     = W + 4 * (size_t)NC;
  float* xv     = W + 5 * (size_t)NC;
  float* xa     = W + 6 * (size_t)NC;
  float* xg     = W + 7 * (size_t)NC;
  float* rb     = W + 8 * (size_t)NC;
  float* wb     = W + 9 * (size_t)NC;
  float* kb     = W + 10 * (size_t)NC;
  float* vb     = W + 11 * (size_t)NC;
  float* ab     = W + 12 * (size_t)NC;
  float* gb     = W + 13 * (size_t)NC;
  float* kkb    = W + 14 * (size_t)NC;
  float* vfirst = W + 15 * (size_t)NC;
  float* yb     = W + 16 * (size_t)NC;
  float* t2     = W + 17 * (size_t)NC;   // LoRA outputs / att pre-proj
  float* h1     = W + 18 * (size_t)NC;   // LoRA hiddens (<= 1024x128)
  float* ffnh   = W + 19 * (size_t)NC;   // 1024x3072 (4*NC)

  // f16 packed-weight arena after the fp32 scratch (23*NC floats)
  _Float16* wh = (_Float16*)(W + 23 * (size_t)NC);
  size_t wofs = 0;
  auto PACK = [&](const float* src, int K, int N) -> const _Float16* {
    int ktiles = K >> 5;
    long ntiles = (N + 127) >> 7;
    long total = ntiles * ktiles * 4096L;
    _Float16* d = wh + wofs;
    wofs += (size_t)total;
    pack_w_kernel<<<(unsigned)((total + 255) / 256), 256, 0, stream>>>(d, src, N, ktiles, total);
    return d;
  };

  // pre-pack all weights to f16 tile order (also halves HBM weight traffic;
  // at ~140MB the full packed set is L2-resident on the 192MB L2)
  const _Float16 *pWr[6], *pWk[6], *pWv[6], *pWo[6], *pW1[6], *pW2[6],
                 *pA1[6], *pA2[6], *pV1[6], *pV2[6], *pG1[6], *pG2[6],
                 *pKf[6], *pVf[6];
  for (int li = 0; li < 6; li++) {
    pWr[li] = PACK(LP(li, P_WR),  CDIM, CDIM);
    pWk[li] = PACK(LP(li, P_WK),  CDIM, CDIM);
    pWv[li] = PACK(LP(li, P_WV),  CDIM, CDIM);
    pWo[li] = PACK(LP(li, P_WO),  CDIM, CDIM);
    pW1[li] = PACK(LP(li, P_W1),  CDIM, 64);
    pW2[li] = PACK(LP(li, P_W2),  64,   CDIM);
    pA1[li] = PACK(LP(li, P_A1),  CDIM, 64);
    pA2[li] = PACK(LP(li, P_A2),  64,   CDIM);
    pV1[li] = PACK(LP(li, P_V1),  CDIM, 32);
    pV2[li] = PACK(LP(li, P_V2),  32,   CDIM);
    pG1[li] = PACK(LP(li, P_G1),  CDIM, 128);
    pG2[li] = PACK(LP(li, P_G2),  128,  CDIM);
    pKf[li] = PACK(LP(li, P_WKF), CDIM, DFFN);
    pVf[li] = PACK(LP(li, P_WVF), DFFN, CDIM);
  }
  const _Float16* pHead = PACK(headW, CDIM, VOCAB);

  const int EW_BLKS = (NC + 255) / 256;

  // embedding + ln0
  embed_kernel<<<NTOK, 256, 0, stream>>>(ids, emb, x);
  ln_kernel<<<NTOK, 256, 0, stream>>>(x, x, LP(0, P_LN0W), LP(0, P_LN0B), 1e-5f);

  for (int li = 0; li < 6; li++) {
    // ---- time mix ----
    ln_kernel<<<NTOK, 256, 0, stream>>>(xn, x, LP(li, P_LN1W), LP(li, P_LN1B), 1e-5f);
    mix6_kernel<<<NTOK, 256, 0, stream>>>(xn,
        LP(li, P_XR), LP(li, P_XW), LP(li, P_XK), LP(li, P_XV), LP(li, P_XA), LP(li, P_XG),
        xr, xw, xk, xv, xa, xg);

    launch_gemm(0, xr, pWr[li], rb, nullptr, NTOK, CDIM, CDIM, stream);
    launch_gemm(0, xk, pWk[li], kb, nullptr, NTOK, CDIM, CDIM, stream);
    launch_gemm(0, xv, pWv[li], vb, nullptr, NTOK, CDIM, CDIM, stream);

    // decay LoRA: tanh(xw@w1)@w2 -> wdecay
    launch_gemm(1, xw, pW1[li], h1, nullptr, NTOK, 64, CDIM, stream);
    launch_gemm(0, h1, pW2[li], t2, nullptr, NTOK, CDIM, 64, stream);
    wdecay_kernel<<<EW_BLKS, 256, 0, stream>>>(wb, t2, LP(li, P_W0), NC);

    // a LoRA: sigmoid(a0 + (xa@a1)@a2)
    launch_gemm(0, xa, pA1[li], h1, nullptr, NTOK, 64, CDIM, stream);
    launch_gemm(0, h1, pA2[li], t2, nullptr, NTOK, CDIM, 64, stream);
    asig_kernel<<<EW_BLKS, 256, 0, stream>>>(ab, t2, LP(li, P_A0), NC);

    // v residual mixing
    if (li == 0) {
      copy_kernel<<<EW_BLKS, 256, 0, stream>>>(vfirst, vb, NC);
    } else {
      launch_gemm(0, xv, pV1[li], h1, nullptr, NTOK, 32, CDIM, stream);
      launch_gemm(0, h1, pV2[li], t2, nullptr, NTOK, CDIM, 32, stream);
      vmix_kernel<<<EW_BLKS, 256, 0, stream>>>(vb, vfirst, t2, LP(li, P_V0), NC);
    }

    // gate: sigmoid(xg@g1)@g2
    launch_gemm(2, xg, pG1[li], h1, nullptr, NTOK, 128, CDIM, stream);
    launch_gemm(0, h1, pG2[li], gb, nullptr, NTOK, CDIM, 128, stream);

    // kk normalize / k rescale (needs ab)
    kk_kernel<<<NTOK * NHD, HSZ, 0, stream>>>(kkb, kb, ab, LP(li, P_KKP), LP(li, P_KAP));

    // sequential scan
    wkv_kernel<<<BB * NHD, HSZ, 0, stream>>>(yb, rb, wb, kb, vb, kkb, ab);

    // groupnorm + bonus + gate -> t2 ; W_o with fused residual into x
    gnbonus_kernel<<<NTOK * NHD, HSZ, 0, stream>>>(t2, yb, rb, kb, vb, gb,
        LP(li, P_LNXW), LP(li, P_LNXB), LP(li, P_RK));
    launch_gemm(0, t2, pWo[li], x, x, NTOK, CDIM, CDIM, stream);

    // ---- channel mix ----
    ln_kernel<<<NTOK, 256, 0, stream>>>(xn, x, LP(li, P_LN2W), LP(li, P_LN2B), 1e-5f);
    mix1_kernel<<<NTOK, 256, 0, stream>>>(xn, LP(li, P_XKF), xr);
    launch_gemm(3, xr, pKf[li], ffnh, nullptr, NTOK, DFFN, CDIM, stream);
    launch_gemm(0, ffnh, pVf[li], x, x, NTOK, CDIM, DFFN, stream);
  }

  // final LN + head projection -> logits (B,T,V) fp32
  ln_kernel<<<NTOK, 256, 0, stream>>>(xn, x, lno_w, lno_b, 1e-5f);
  launch_gemm(0, xn, pHead, (float*)d_out, nullptr, NTOK, VOCAB, CDIM, stream);
}
